// AttentionBlock_4148938408587
// MI455X (gfx1250) — compile-verified, matrix operand order checked by reading, not run on hardware
//
#include <hip/hip_runtime.h>
#include <hip/hip_bf16.h>
#include <stdint.h>

#define BATCH 8
#define CCH   128
#define NN    4096   // 64*64
#define PDIM  66     // padded spatial edge (64 + halo)

typedef unsigned int   u32;
typedef unsigned short u16;
typedef __attribute__((ext_vector_type(4)))  u32    v4u;
typedef __attribute__((ext_vector_type(8)))  u32    v8u;
typedef __attribute__((ext_vector_type(16))) __bf16 v16bf;
typedef __attribute__((ext_vector_type(8)))  float  v8f;

// fp32 -> bf16 round-to-nearest-even
static __device__ __forceinline__ u16 f2bf(float f) {
  u32 u = __float_as_uint(f);
  u32 r = u + 0x7FFFu + ((u >> 16) & 1u);
  return (u16)(r >> 16);
}

static __device__ __forceinline__ v16bf bf_frag(uint4 lo, uint4 hi) {
  v8u t;
  t[0] = lo.x; t[1] = lo.y; t[2] = lo.z; t[3] = lo.w;
  t[4] = hi.x; t[5] = hi.y; t[6] = hi.z; t[7] = hi.w;
  return __builtin_bit_cast(v16bf, t);
}
static __device__ __forceinline__ v8f wmma_bf16(v16bf a, v16bf b, v8f c) {
  return __builtin_amdgcn_wmma_f32_16x16x32_bf16(false, a, false, b, (short)0, c,
                                                 false, false);
}

// ---------------- pack kernels ----------------

// Wq/Wk/Wv: [co][ci][3][3] f32 -> [tap][co][ci] bf16 ; Wo: [co][ci] bf16
__global__ __launch_bounds__(256) void pack_w_kernel(
    const float* __restrict__ Wq, const float* __restrict__ Wk,
    const float* __restrict__ Wv, const float* __restrict__ Wo,
    u16* __restrict__ wq, u16* __restrict__ wk, u16* __restrict__ wv,
    u16* __restrict__ wo) {
  int idx = blockIdx.x * 256 + threadIdx.x;
  if (idx < CCH * CCH * 9) {
    int tap = idx % 9;
    int ci  = (idx / 9) % CCH;
    int co  = idx / (9 * CCH);
    size_t dst = ((size_t)tap * CCH + co) * CCH + ci;
    wq[dst] = f2bf(Wq[idx]);
    wk[dst] = f2bf(Wk[idx]);
    wv[dst] = f2bf(Wv[idx]);
  }
  if (idx < CCH * CCH) wo[idx] = f2bf(Wo[idx]);
}

// x: [b][c][64][64] f32 -> xP: [b][66][66][c] bf16 (zero halo -> branch-free conv)
__global__ __launch_bounds__(256) void pack_x_kernel(const float* __restrict__ x,
                                                     u16* __restrict__ xP) {
  size_t idx = (size_t)blockIdx.x * 256 + threadIdx.x;  // over B*66*66*C
  int c  = (int)(idx % CCH);
  int px = (int)((idx / CCH) % PDIM);
  int py = (int)((idx / (CCH * PDIM)) % PDIM);
  int b  = (int)(idx / ((size_t)CCH * PDIM * PDIM));
  u16 v = 0;
  if (px >= 1 && px <= 64 && py >= 1 && py <= 64) {
    int n = (py - 1) * 64 + (px - 1);
    v = f2bf(x[((size_t)b * CCH + c) * NN + n]);
  }
  xP[idx] = v;
}

// ---------------- 3x3 conv as 9 shifted GEMMs (branch-free via padded input) --
__global__ __launch_bounds__(256) void conv3_kernel(
    const u16* __restrict__ xP, const u16* __restrict__ wP,
    const float* __restrict__ bias, u16* __restrict__ out, int transposed) {
  const int lane = threadIdx.x & 31;
  const int wid  = threadIdx.x >> 5;
  const int col  = lane & 15;
  const int half = lane >> 4;
  const int koff = half * 8;
  const int b  = blockIdx.z;
  const int m0 = blockIdx.y * 16;
  const int n0 = (blockIdx.x * 8 + wid) * 16;

  const int n  = n0 + col;
  const int y  = n >> 6;
  const int xc = n & 63;

  v8f acc = {};
#pragma unroll
  for (int tap = 0; tap < 9; ++tap) {
    const int dy = tap / 3, dx = tap % 3;          // padded coords: +0..2
    const int pp = (y + dy) * PDIM + (xc + dx);
    const u16* bbase = xP + ((size_t)b * PDIM * PDIM + pp) * CCH;
    const u16* abase = wP + ((size_t)tap * CCH + (m0 + col)) * CCH;
#pragma unroll
    for (int kc = 0; kc < CCH; kc += 32) {
      const u16* ap = abase + kc + koff;
      const u16* bp = bbase + kc + koff;
      acc = wmma_bf16(bf_frag(*(const uint4*)ap, *(const uint4*)(ap + 16)),
                      bf_frag(*(const uint4*)bp, *(const uint4*)(bp + 16)), acc);
    }
  }
#pragma unroll
  for (int r = 0; r < 8; ++r) {
    const int m = m0 + r + 8 * half;
    float v = acc[r] + bias[m];
    if (transposed) out[((size_t)b * NN + n) * CCH + m] = f2bf(v);   // [b][n][c]
    else            out[((size_t)b * CCH + m) * NN + n] = f2bf(v);   // [b][c][n]
  }
}

// ---------------- flash attention ----------------
// 4 waves/WG, 16 query rows/wave. K tiles staged once per WG into LDS via the
// Tensor Data Mover (pad_enable inserts 16B per 256B row -> conflict-free
// ds_load_b128 fragment reads, row stride 272B).
#define KROW 136  // u16 elements per padded LDS row (272 bytes)

__global__ __launch_bounds__(128) void flash_kernel(
    const u16* __restrict__ qT, const u16* __restrict__ kT,
    const u16* __restrict__ vG, u16* __restrict__ oT) {
  __shared__ __align__(16) u16 skt[32][KROW];   // TDM-staged K tile (8.5 KB)
  __shared__ __align__(16) u16 sm[4][16][40];   // per-wave P tile
  const int lane = threadIdx.x & 31;
  const int wid  = threadIdx.x >> 5;
  const int col  = lane & 15;
  const int half = lane >> 4;
  const int koff = half * 8;
  const int b  = blockIdx.y;
  const int i0 = (blockIdx.x * 4 + wid) * 16;
  const float scale = 0.088388347648318447f;  // 1/sqrt(128)

  v16bf qa[4];
#pragma unroll
  for (int kc = 0; kc < 4; ++kc) {
    const u16* p = qT + ((size_t)b * NN + i0 + col) * CCH + kc * 32 + koff;
    qa[kc] = bf_frag(*(const uint4*)p, *(const uint4*)(p + 16));
  }

  const u32 lds_base = (u32)(size_t)(const void*)&skt[0][0];

  v8f zero8 = {};
  v8f acc[8];
  float mrow[8], lsum[8];
#pragma unroll
  for (int ct = 0; ct < 8; ++ct) acc[ct] = zero8;
#pragma unroll
  for (int r = 0; r < 8; ++r) { mrow[r] = -3.0e38f; lsum[r] = 0.0f; }

  for (int j0 = 0; j0 < NN; j0 += 32) {
    __syncthreads();  // all waves done with previous K tile
    if (wid == 0) {
      // TDM descriptor: 2D tile, 32 rows x 32 elems of 8B (=256B/row),
      // contiguous in memory, +4 dwords LDS pad per 64 dwords (per row).
      unsigned long long ga =
          (unsigned long long)(size_t)(kT + ((size_t)b * NN + j0) * CCH);
      v4u g0;
      g0[0] = 1u;                                  // count=1, user descriptor
      g0[1] = lds_base;                            // lds_addr
      g0[2] = (u32)ga;                             // global_addr[31:0]
      g0[3] = ((u32)(ga >> 32) & 0x01FFFFFFu) | (2u << 30);  // addr[56:32]|type=2
      v8u g1;
      g1[0] = (3u << 16)      // data_size = 8B
            | (1u << 20)      // pad_enable
            | (5u << 22)      // pad_interval = 64 dwords (one 256B row)
            | (3u << 25);     // pad_amount = 4 dwords (16B)
      g1[1] = 32u << 16;      // tensor_dim0 = 32 (8B units)
      g1[2] = 32u << 16;      // tensor_dim1 = 32 rows
      g1[3] = 32u << 16;      // tile_dim0 = 32
      g1[4] = 32u;            // tile_dim1 = 32, tile_dim2 = 0
      g1[5] = 32u;            // tensor_dim0_stride = 32 (contiguous rows)
      g1[6] = 0u;
      g1[7] = 0u;
      v4u gz = {0u, 0u, 0u, 0u};
      asm volatile("tensor_load_to_lds %0, %1, %2, %3"
                   :: "s"(g0), "s"(g1), "s"(gz), "s"(gz) : "memory");
      __builtin_amdgcn_s_wait_tensorcnt(0);
    }
    __syncthreads();  // K tile visible to all waves

    v8f s0 = {}, s1 = {};
#pragma unroll
    for (int kc = 0; kc < 4; ++kc) {
      const u16* p = &skt[col][kc * 32 + koff];
      s0 = wmma_bf16(qa[kc], bf_frag(*(const uint4*)p, *(const uint4*)(p + 16)), s0);
    }
#pragma unroll
    for (int kc = 0; kc < 4; ++kc) {
      const u16* p = &skt[col + 16][kc * 32 + koff];
      s1 = wmma_bf16(qa[kc], bf_frag(*(const uint4*)p, *(const uint4*)(p + 16)), s1);
    }
    if (j0 + 32 < NN)
      __builtin_prefetch(vG + ((size_t)b * CCH + col) * NN + j0 + 32, 0, 1);

#pragma unroll
    for (int r = 0; r < 8; ++r) {
      float v0 = s0[r] * scale, v1 = s1[r] * scale;
      float t = fmaxf(v0, v1);
#pragma unroll
      for (int off = 1; off < 16; off <<= 1) t = fmaxf(t, __shfl_xor(t, off, 32));
      float mn = fmaxf(mrow[r], t);
      float al = __expf(mrow[r] - mn);
      float p0 = __expf(v0 - mn);
      float p1 = __expf(v1 - mn);
      float rs = p0 + p1;
#pragma unroll
      for (int off = 1; off < 16; off <<= 1) rs += __shfl_xor(rs, off, 32);
      lsum[r] = lsum[r] * al + rs;
      mrow[r] = mn;
#pragma unroll
      for (int ct = 0; ct < 8; ++ct) acc[ct][r] *= al;
      const int row = r + 8 * half;
      sm[wid][row][col]      = f2bf(p0);
      sm[wid][row][col + 16] = f2bf(p1);
    }
    // C-layout -> A-fragment layout via LDS (same-wave DS ops are in-order)
    const u16* sp = &sm[wid][col][0];
    v16bf pf = bf_frag(*(const uint4*)(sp + koff), *(const uint4*)(sp + koff + 16));
#pragma unroll
    for (int ct = 0; ct < 8; ++ct) {
      const u16* vp = vG + ((size_t)b * CCH + ct * 16 + col) * NN + j0 + koff;
      v16bf vf = bf_frag(*(const uint4*)vp, *(const uint4*)(vp + 16));
      acc[ct] = wmma_bf16(pf, vf, acc[ct]);
    }
  }
#pragma unroll
  for (int r = 0; r < 8; ++r) {
    float inv = 1.0f / lsum[r];
    const int row = i0 + r + 8 * half;
#pragma unroll
    for (int ct = 0; ct < 8; ++ct) {
      oT[((size_t)b * NN + row) * CCH + ct * 16 + col] = f2bf(acc[ct][r] * inv);
    }
  }
}

// ---------------- 1x1 conv + bias + residual ----------------
__global__ __launch_bounds__(256) void out_kernel(
    const u16* __restrict__ oT, const u16* __restrict__ woP,
    const float* __restrict__ bo, const float* __restrict__ x,
    float* __restrict__ out) {
  const int lane = threadIdx.x & 31;
  const int wid  = threadIdx.x >> 5;
  const int col  = lane & 15;
  const int half = lane >> 4;
  const int koff = half * 8;
  const int b  = blockIdx.z;
  const int m0 = blockIdx.y * 16;
  const int n0 = (blockIdx.x * 8 + wid) * 16;

  v8f acc = {};
#pragma unroll
  for (int kc = 0; kc < CCH; kc += 32) {
    const u16* ap = woP + (size_t)(m0 + col) * CCH + kc + koff;
    const u16* bp = oT + ((size_t)b * NN + n0 + col) * CCH + kc + koff;
    acc = wmma_bf16(bf_frag(*(const uint4*)ap, *(const uint4*)(ap + 16)),
                    bf_frag(*(const uint4*)bp, *(const uint4*)(bp + 16)), acc);
  }
#pragma unroll
  for (int r = 0; r < 8; ++r) {
    const int m = m0 + r + 8 * half;
    const size_t idx = ((size_t)b * CCH + m) * NN + (n0 + col);
    out[idx] = acc[r] + bo[m] + x[idx];
  }
}

extern "C" void kernel_launch(void* const* d_in, const int* in_sizes, int n_in,
                              void* d_out, int out_size, void* d_ws, size_t ws_size,
                              hipStream_t stream) {
  (void)in_sizes; (void)n_in; (void)out_size; (void)ws_size;
  const float* x  = (const float*)d_in[0];
  const float* Wq = (const float*)d_in[1];
  const float* bq = (const float*)d_in[2];
  const float* Wk = (const float*)d_in[3];
  const float* bk = (const float*)d_in[4];
  const float* Wv = (const float*)d_in[5];
  const float* bv = (const float*)d_in[6];
  const float* Wo = (const float*)d_in[7];
  const float* bo = (const float*)d_in[8];

  char* ws = (char*)d_ws;
  size_t o = 0;
  const size_t psz = (size_t)BATCH * PDIM * PDIM * CCH * sizeof(u16);  // padded x
  const size_t tsz = (size_t)BATCH * NN * CCH * sizeof(u16);           // 8 MB each
  u16* xP = (u16*)(ws + o); o += psz;
  u16* qT = (u16*)(ws + o); o += tsz;
  u16* kT = (u16*)(ws + o); o += tsz;
  u16* vG = (u16*)(ws + o); o += tsz;
  u16* oT = (u16*)(ws + o); o += tsz;
  u16* wq = (u16*)(ws + o); o += (size_t)9 * CCH * CCH * sizeof(u16);
  u16* wk = (u16*)(ws + o); o += (size_t)9 * CCH * CCH * sizeof(u16);
  u16* wv = (u16*)(ws + o); o += (size_t)9 * CCH * CCH * sizeof(u16);
  u16* wo = (u16*)(ws + o); o += (size_t)CCH * CCH * sizeof(u16);

  pack_w_kernel<<<(CCH * CCH * 9 + 255) / 256, 256, 0, stream>>>(
      Wq, Wk, Wv, Wo, wq, wk, wv, wo);
  pack_x_kernel<<<((size_t)BATCH * PDIM * PDIM * CCH + 255) / 256, 256, 0,
                  stream>>>(x, xP);

  dim3 g(NN / (16 * 8), CCH / 16, BATCH);  // (32, 8, 8)
  conv3_kernel<<<g, 256, 0, stream>>>(xP, wq, bq, qT, 1);  // qT: [b][n][c]
  conv3_kernel<<<g, 256, 0, stream>>>(xP, wk, bk, kT, 1);  // kT: [b][n][c]
  conv3_kernel<<<g, 256, 0, stream>>>(xP, wv, bv, vG, 0);  // vG: [b][c][n]

  flash_kernel<<<dim3(NN / 64, BATCH), 128, 0, stream>>>(qT, kT, vG, oT);

  out_kernel<<<g, 256, 0, stream>>>(oT, wo, bo, x, (float*)d_out);
}
